// AttentionGate3D_1279900254705
// MI455X (gfx1250) — matrix operand order repaired and checked by reading, not compiled, not run on hardware
//
#include <hip/hip_runtime.h>

// AttentionGate3D for MI455X (gfx1250, wave32, WMMA bf16).
// out[b,c,p] = x[b,c,p] * sigmoid( Wpsi . relu( [Wg|Wx]@[g;x](p) + bg+bx ) + bpsi )

namespace {
constexpr int C_IN  = 512;              // input channels per tensor
constexpr int C_INT = 256;              // intermediate channels (GEMM M)
constexpr int VOXB  = 32 * 64 * 64;     // voxels per batch = 131072 (channel stride)
constexpr int NTILE = 128;              // voxels per block (GEMM N tile)
constexpr int KTOT  = 2 * C_IN;         // 1024: k<512 -> g, k>=512 -> x
constexpr int KC    = 64;               // K chunk staged in LDS
constexpr int BLKPB = VOXB / NTILE;     // 1024 blocks per batch
}

typedef __attribute__((ext_vector_type(16))) __bf16 v16bf;
typedef __attribute__((ext_vector_type(8)))  __bf16 v8bf;
typedef __attribute__((ext_vector_type(2)))  __bf16 v2bf;
typedef __attribute__((ext_vector_type(8)))  float  v8f;

static __device__ __forceinline__ unsigned f2bf2(float lo, float hi) {
#if __has_builtin(__builtin_amdgcn_cvt_pk_bf16_f32)
    // single VALU op if the toolchain exposes v_cvt_pk_bf16_f32
    union { v2bf v; unsigned u; } r;
    r.v = __builtin_amdgcn_cvt_pk_bf16_f32(lo, hi);
    return r.u;
#else
    // 3 VALU ops: round-to-nearest via +0x8000, then one v_perm_b32 to pack
    // the two high halves: D = {hi[31:16], lo[31:16]}
    const unsigned ra = __float_as_uint(lo) + 0x8000u;
    const unsigned rb = __float_as_uint(hi) + 0x8000u;
    return __builtin_amdgcn_perm(rb, ra, 0x07060302u);
#endif
}

__global__ __launch_bounds__(256) void attn_gate3d_kernel(
    const float* __restrict__ g,  const float* __restrict__ x,
    const float* __restrict__ Wg, const float* __restrict__ bg,
    const float* __restrict__ Wx, const float* __restrict__ bx,
    const float* __restrict__ Wpsi, const float* __restrict__ bpsi,
    float* __restrict__ out)
{
    __shared__ __align__(32) unsigned short Alds[C_INT * KC]; // 32 KB, [m][k] bf16
    __shared__ __align__(32) unsigned short Blds[NTILE * KC]; // 16 KB, [n][k] bf16
    __shared__ float bsumL[C_INT];   // bg + bx
    __shared__ float wpsiL[C_INT];   // Wpsi
    __shared__ float psiL[NTILE];    // per-voxel psi accumulator
    __shared__ float attnL[NTILE];   // sigmoid result

    const int t    = threadIdx.x;
    const int lane = t & 31;
    const int wave = t >> 5;         // 0..7, owns M rows [32*wave, 32*wave+32)
    const int half = lane >> 4;      // 0|1
    const int l16  = lane & 15;

    const int  batch = blockIdx.x / BLKPB;
    const int  p0    = (blockIdx.x % BLKPB) * NTILE;
    const long base  = (long)batch * C_IN * VOXB + p0;
    const float* gB  = g + base;
    const float* xB  = x + base;

    bsumL[t] = bg[t] + bx[t];
    wpsiL[t] = Wpsi[t];
    if (t < NTILE) psiL[t] = 0.0f;   // ordered before atomics by loop barriers

    v8f acc[2][8] = {};              // 2 M-tiles x 8 N-tiles, f32 16x16 each

    for (int kc = 0; kc < KTOT; kc += KC) {
        const float* Wsrc = (kc < C_IN) ? (Wg + kc) : (Wx + (kc - C_IN));
        const float* Bsrc = (kc < C_IN) ? (gB + (long)kc * VOXB)
                                        : (xB + (long)(kc - C_IN) * VOXB);
        // ---- stage A: weights (256 x 64) fp32 -> bf16 LDS, coalesced 256B/wave
        {
            const int k2 = (t & 31) * 2;
            const int m0 = t >> 5;
            #pragma unroll
            for (int i = 0; i < 32; ++i) {
                const int m = m0 + i * 8;
                const float2 v = *(const float2*)&Wsrc[m * C_IN + k2];
                *(unsigned*)&Alds[m * KC + k2] = f2bf2(v.x, v.y);
            }
        }
        // ---- stage B: inputs (64 x 128) fp32 -> bf16 LDS [n][k], coalesced 128B/load
        {
            const int n  = t & 127;
            const int kq = (t >> 7) * 32;    // 0 or 32
            unsigned pk[16];
            #pragma unroll
            for (int j = 0; j < 16; ++j) {
                const float a0 = Bsrc[(long)(kq + 2 * j)     * VOXB + n];
                const float a1 = Bsrc[(long)(kq + 2 * j + 1) * VOXB + n];
                pk[j] = f2bf2(a0, a1);
            }
            uint4* dst = (uint4*)&Blds[n * KC + kq];
            dst[0] = make_uint4(pk[0],  pk[1],  pk[2],  pk[3]);
            dst[1] = make_uint4(pk[4],  pk[5],  pk[6],  pk[7]);
            dst[2] = make_uint4(pk[8],  pk[9],  pk[10], pk[11]);
            dst[3] = make_uint4(pk[12], pk[13], pk[14], pk[15]);
        }
        // ---- prefetch next chunk's input rows (global_prefetch_b8)
        if (kc + KC < KTOT) {
            const int kc2 = kc + KC;
            const float* Pn = (kc2 < C_IN) ? (gB + (long)kc2 * VOXB)
                                           : (xB + (long)(kc2 - C_IN) * VOXB);
            __builtin_prefetch(&Pn[(long)((t >> 7) * 32) * VOXB + (t & 127)], 0, 1);
        }
        __syncthreads();

        // ---- 2 WMMA k-steps of 32 over this chunk
        #pragma unroll
        for (int ks = 0; ks < 2; ++ks) {
            const int akb = ks * 32 + half * 8;   // A: lanes<16 K base 0, lanes>=16 base 8
            const int bkb = ks * 32 + half * 8;   // B: same K layout as A (base, base+16)
            union { v16bf v; v8bf h[2]; } a[2];
            #pragma unroll
            for (int mt = 0; mt < 2; ++mt) {
                const int row = (wave * 2 + mt) * 16 + l16;   // A row M = l16 (both halves)
                a[mt].h[0] = *(const v8bf*)&Alds[row * KC + akb];        // K base..base+7
                a[mt].h[1] = *(const v8bf*)&Alds[row * KC + akb + 16];   // K base+16..+23
            }
            #pragma unroll
            for (int nt = 0; nt < 8; ++nt) {
                union { v16bf v; v8bf h[2]; } bu;
                bu.h[0] = *(const v8bf*)&Blds[(nt * 16 + l16) * KC + bkb];
                bu.h[1] = *(const v8bf*)&Blds[(nt * 16 + l16) * KC + bkb + 16];
                const v16bf b = bu.v;
                acc[0][nt] = __builtin_amdgcn_wmma_f32_16x16x32_bf16(
                    false, a[0].v, false, b, (short)0, acc[0][nt], false, false);
                acc[1][nt] = __builtin_amdgcn_wmma_f32_16x16x32_bf16(
                    false, a[1].v, false, b, (short)0, acc[1][nt], false, false);
            }
        }
        __syncthreads();
    }

    // ---- epilogue: bias + relu + Wpsi dot, reduced across waves in LDS
    {
        float part[8];
        #pragma unroll
        for (int nt = 0; nt < 8; ++nt) part[nt] = 0.0f;
        #pragma unroll
        for (int mt = 0; mt < 2; ++mt) {
            const int mb = (wave * 2 + mt) * 16 + half * 8;  // C/D: VGPR r -> M = mb + r
            #pragma unroll
            for (int r = 0; r < 8; ++r) {
                const float bsum = bsumL[mb + r];
                const float wp   = wpsiL[mb + r];
                #pragma unroll
                for (int nt = 0; nt < 8; ++nt)
                    part[nt] += wp * fmaxf(acc[mt][nt][r] + bsum, 0.0f);
            }
        }
        #pragma unroll
        for (int nt = 0; nt < 8; ++nt)
            atomicAdd(&psiL[nt * 16 + l16], part[nt]);       // ds_add_f32
    }
    __syncthreads();
    if (t < NTILE) {
        const float s = psiL[t] + bpsi[0];
        attnL[t] = 1.0f / (1.0f + __expf(-s));
    }
    __syncthreads();

    // ---- out = x * attn, coalesced 128B per wave per store
    {
        const int n  = t & 127;
        const int c0 = t >> 7;                // 0 or 1
        const float a = attnL[n];
        const float* xc = xB + n;
        float* oc = out + base + n;
        #pragma unroll 8
        for (int c = c0; c < C_IN; c += 2)
            oc[(long)c * VOXB] = xc[(long)c * VOXB] * a;
    }
}

extern "C" void kernel_launch(void* const* d_in, const int* in_sizes, int n_in,
                              void* d_out, int out_size, void* d_ws, size_t ws_size,
                              hipStream_t stream) {
    (void)n_in; (void)out_size; (void)d_ws; (void)ws_size;
    const float* g    = (const float*)d_in[0];
    const float* x    = (const float*)d_in[1];
    const float* Wg   = (const float*)d_in[2];
    const float* bg   = (const float*)d_in[3];
    const float* Wx   = (const float*)d_in[4];
    const float* bx   = (const float*)d_in[5];
    const float* Wpsi = (const float*)d_in[6];
    const float* bpsi = (const float*)d_in[7];
    float* out = (float*)d_out;

    const int batches = in_sizes[0] / (C_IN * VOXB);   // 2
    dim3 grid(batches * BLKPB);                        // 2048 blocks
    attn_gate3d_kernel<<<grid, 256, 0, stream>>>(g, x, Wg, bg, Wx, bx, Wpsi, bpsi, out);
}
